// SparseMoE_60533269070475
// MI455X (gfx1250) — matrix operand order repaired and checked by reading, compiled, not run on hardware
//
#include <hip/hip_runtime.h>
#include <math.h>

// ---------------- problem constants (fixed by the reference) ----------------
#define T_TOK   4096      // B*S tokens
#define DM      1024      // d_model
#define DFF     4096      // d_ff
#define NE      8         // experts
#define TOPK    2

typedef _Float16 half8  __attribute__((ext_vector_type(8)));
typedef _Float16 half16 __attribute__((ext_vector_type(16)));
typedef float    v8f    __attribute__((ext_vector_type(8)));

// ---------------------------------------------------------------------------
// Fragment loaders in contiguous k order (not the ISA table order; valid since A and B match).
// A (16x32 f16): lane L -> row M = L%16 ; elements 0..15 = K = kb+0..15,
//                kb = 16*(L/16) (contiguous, same as B).
// B (32x16 f16): lane L -> col N = L%16 ; elements 0..15 = K = 16*(L/16)+0..15
//                (contiguous), matching the sparse-B table pattern.
// Both loaded from row-major [row, K] f16 arrays.
// ---------------------------------------------------------------------------
__device__ __forceinline__ half16 load_fragA(const _Float16* p /* = row + k0 + 16*(L/16) */) {
    half8 lo = *(const half8*)(p);
    half8 hi = *(const half8*)(p + 8);
    return __builtin_shufflevector(lo, hi, 0,1,2,3,4,5,6,7,8,9,10,11,12,13,14,15);
}
__device__ __forceinline__ half16 load_fragB(const _Float16* p /* = row + k0 + 16*(L/16) */) {
    return *(const half16*)(p);
}

__device__ __forceinline__ float silu_f(float v) {
    return v / (1.0f + __expf(-v));
}

// ---------------- kernel 0: zero routing counters --------------------------
__global__ void moe_zero_kernel(int* counts, float* probsum) {
    int i = threadIdx.x;
    if (i < NE) { counts[i] = 0; probsum[i] = 0.0f; }
}

// ---------------- kernel 1: fp32 -> f16 convert ----------------------------
__global__ void moe_cvt_kernel(const float* __restrict__ in, _Float16* __restrict__ out, long n) {
    long i = (long)blockIdx.x * blockDim.x + threadIdx.x;
    long stride = (long)gridDim.x * blockDim.x;
    for (; i < n; i += stride) out[i] = (_Float16)in[i];
}

// ---------------- kernel 2: gating / routing -------------------------------
// One block per token: logits over 8 experts, top-2 softmax routing,
// gate-prob accumulation for the load-balance loss.
__global__ void moe_gating_kernel(const float* __restrict__ x,
                                  const float* __restrict__ gw,
                                  int* __restrict__ counts,
                                  float* __restrict__ probsum,
                                  int* __restrict__ tokIdx,
                                  int* __restrict__ tokSlot,
                                  float* __restrict__ tokWgt) {
    __shared__ float sd[256 * NE];
    const int t   = blockIdx.x;
    const int tid = threadIdx.x;

    float p[NE];
#pragma unroll
    for (int e = 0; e < NE; ++e) p[e] = 0.0f;

    const float* xr = x + (long)t * DM;
    for (int k = tid; k < DM; k += 256) {
        float xv = xr[k];
#pragma unroll
        for (int e = 0; e < NE; ++e) p[e] += xv * gw[e * DM + k];
    }
#pragma unroll
    for (int e = 0; e < NE; ++e) sd[tid * NE + e] = p[e];
    __syncthreads();

    for (int s = 128; s > 0; s >>= 1) {
        if (tid < s) {
#pragma unroll
            for (int e = 0; e < NE; ++e) sd[tid * NE + e] += sd[(tid + s) * NE + e];
        }
        __syncthreads();
    }

    if (tid == 0) {
        float l[NE];
#pragma unroll
        for (int e = 0; e < NE; ++e) l[e] = sd[e];
        // top-2 (first index wins ties, matching lax.top_k)
        int i0 = 0;
#pragma unroll
        for (int e = 1; e < NE; ++e) if (l[e] > l[i0]) i0 = e;
        int i1 = (i0 == 0) ? 1 : 0;
#pragma unroll
        for (int e = 0; e < NE; ++e) if (e != i0 && l[e] > l[i1]) i1 = e;
        // softmax over the two kept logits
        float m  = l[i0];
        float e1 = __expf(l[i1] - m);
        float den = 1.0f + e1;
        float w0 = 1.0f / den;
        float w1 = e1 / den;
        // push into per-expert lists
        int p0 = atomicAdd(&counts[i0], 1);
        tokIdx[i0 * T_TOK + p0] = t; tokSlot[i0 * T_TOK + p0] = 0; tokWgt[i0 * T_TOK + p0] = w0;
        int p1 = atomicAdd(&counts[i1], 1);
        tokIdx[i1 * T_TOK + p1] = t; tokSlot[i1 * T_TOK + p1] = 1; tokWgt[i1 * T_TOK + p1] = w1;
        // full softmax for lb_loss
        float se = 0.0f;
        float pe[NE];
#pragma unroll
        for (int e = 0; e < NE; ++e) { pe[e] = __expf(l[e] - m); se += pe[e]; }
        float inv = 1.0f / se;
#pragma unroll
        for (int e = 0; e < NE; ++e) atomicAdd(&probsum[e], pe[e] * inv);
    }
}

// ---------------- kernel 3: GEMM1  h = silu(Xg @ W1e^T) --------------------
// 128-thread block = 4 waves; each wave computes a 16(M tokens) x 64(N=f)
// tile of a shared 64-row stripe, so all 4 waves stream identical B rows
// (WGP$/L2 friendly). K-loop is manually double-buffered so the global
// b128 loads for k+32 are in flight while the 4 WMMAs for k execute.
__global__ void moe_gemm1_kernel(const _Float16* __restrict__ xh,
                                 const _Float16* __restrict__ w1h,
                                 _Float16* __restrict__ h,
                                 const int* __restrict__ counts,
                                 const int* __restrict__ tokIdx,
                                 const int* __restrict__ tokSlot) {
    const int e    = blockIdx.z;
    const int cnt  = counts[e];
    const int wave = threadIdx.x >> 5;
    const int row0 = (blockIdx.y * 4 + wave) * 16;
    if (row0 >= cnt) return;                       // wave-uniform exit, no barriers used
    const int n0   = blockIdx.x * 64;

    const int lane  = threadIdx.x & 31;            // wave32
    const int lrow  = lane & 15;
    const int khalf = lane >> 4;                   // 0/1
    const int kbA   = khalf * 16;                  // A-frag K base within 32
    const int kbB   = khalf * 16;                  // B-frag K base within 32

    int arow = row0 + lrow; if (arow >= cnt) arow = cnt - 1;
    const int  atok = tokIdx[e * T_TOK + arow];
    const _Float16* aptr = xh + (long)atok * DM + kbA;

    const _Float16* wbase = w1h + (long)e * DFF * DM;
    const _Float16* b0 = wbase + (long)(n0 + 0  + lrow) * DM + kbB;
    const _Float16* b1 = wbase + (long)(n0 + 16 + lrow) * DM + kbB;
    const _Float16* b2 = wbase + (long)(n0 + 32 + lrow) * DM + kbB;
    const _Float16* b3 = wbase + (long)(n0 + 48 + lrow) * DM + kbB;

    v8f acc0 = {}, acc1 = {}, acc2 = {}, acc3 = {};
    // prologue: fragments for k=0
    half16 a  = load_fragA(aptr);
    half16 f0 = load_fragB(b0);
    half16 f1 = load_fragB(b1);
    half16 f2 = load_fragB(b2);
    half16 f3 = load_fragB(b3);
    for (int k0 = 0; k0 < DM - 32; k0 += 32) {
        // issue next K-block loads before consuming current fragments
        half16 an = load_fragA(aptr + k0 + 32);
        half16 g0 = load_fragB(b0 + k0 + 32);
        half16 g1 = load_fragB(b1 + k0 + 32);
        half16 g2 = load_fragB(b2 + k0 + 32);
        half16 g3 = load_fragB(b3 + k0 + 32);
        if (k0 + 64 < DM) {
            __builtin_prefetch(b0 + k0 + 64, 0, 1);
            __builtin_prefetch(b2 + k0 + 64, 0, 1);
        }
        acc0 = __builtin_amdgcn_wmma_f32_16x16x32_f16(false, a, false, f0, (short)0, acc0, false, false);
        acc1 = __builtin_amdgcn_wmma_f32_16x16x32_f16(false, a, false, f1, (short)0, acc1, false, false);
        acc2 = __builtin_amdgcn_wmma_f32_16x16x32_f16(false, a, false, f2, (short)0, acc2, false, false);
        acc3 = __builtin_amdgcn_wmma_f32_16x16x32_f16(false, a, false, f3, (short)0, acc3, false, false);
        a = an; f0 = g0; f1 = g1; f2 = g2; f3 = g3;
    }
    // epilogue K-step
    acc0 = __builtin_amdgcn_wmma_f32_16x16x32_f16(false, a, false, f0, (short)0, acc0, false, false);
    acc1 = __builtin_amdgcn_wmma_f32_16x16x32_f16(false, a, false, f1, (short)0, acc1, false, false);
    acc2 = __builtin_amdgcn_wmma_f32_16x16x32_f16(false, a, false, f2, (short)0, acc2, false, false);
    acc3 = __builtin_amdgcn_wmma_f32_16x16x32_f16(false, a, false, f3, (short)0, acc3, false, false);

    // D layout: VGPR r -> M = r + 8*khalf, N = lane%16
#pragma unroll
    for (int r = 0; r < 8; ++r) {
        int row = row0 + r + khalf * 8;
        if (row < cnt) {
            int tk = tokIdx[e * T_TOK + row];
            int sl = tokSlot[e * T_TOK + row];
            long base = ((long)tk * TOPK + sl) * DFF + n0 + lrow;
            h[base +  0] = (_Float16)silu_f(acc0[r]);
            h[base + 16] = (_Float16)silu_f(acc1[r]);
            h[base + 32] = (_Float16)silu_f(acc2[r]);
            h[base + 48] = (_Float16)silu_f(acc3[r]);
        }
    }
}

// ---------------- kernel 4: GEMM2  eo = wgt * (Hg @ W2e^T) -----------------
__global__ void moe_gemm2_kernel(const _Float16* __restrict__ h,
                                 const _Float16* __restrict__ w2h,
                                 float* __restrict__ eo,
                                 const int* __restrict__ counts,
                                 const int* __restrict__ tokIdx,
                                 const int* __restrict__ tokSlot,
                                 const float* __restrict__ tokWgt) {
    const int e    = blockIdx.z;
    const int cnt  = counts[e];
    const int wave = threadIdx.x >> 5;
    const int row0 = (blockIdx.y * 4 + wave) * 16;
    if (row0 >= cnt) return;
    const int n0   = blockIdx.x * 64;

    const int lane  = threadIdx.x & 31;
    const int lrow  = lane & 15;
    const int khalf = lane >> 4;
    const int kbA   = khalf * 16;
    const int kbB   = khalf * 16;

    int arow = row0 + lrow; if (arow >= cnt) arow = cnt - 1;
    const int atok = tokIdx[e * T_TOK + arow];
    const int aslt = tokSlot[e * T_TOK + arow];
    const _Float16* aptr = h + ((long)atok * TOPK + aslt) * DFF + kbA;

    const _Float16* wbase = w2h + (long)e * DM * DFF;
    const _Float16* b0 = wbase + (long)(n0 + 0  + lrow) * DFF + kbB;
    const _Float16* b1 = wbase + (long)(n0 + 16 + lrow) * DFF + kbB;
    const _Float16* b2 = wbase + (long)(n0 + 32 + lrow) * DFF + kbB;
    const _Float16* b3 = wbase + (long)(n0 + 48 + lrow) * DFF + kbB;

    v8f acc0 = {}, acc1 = {}, acc2 = {}, acc3 = {};
    half16 a  = load_fragA(aptr);
    half16 f0 = load_fragB(b0);
    half16 f1 = load_fragB(b1);
    half16 f2 = load_fragB(b2);
    half16 f3 = load_fragB(b3);
    for (int k0 = 0; k0 < DFF - 32; k0 += 32) {
        half16 an = load_fragA(aptr + k0 + 32);
        half16 g0 = load_fragB(b0 + k0 + 32);
        half16 g1 = load_fragB(b1 + k0 + 32);
        half16 g2 = load_fragB(b2 + k0 + 32);
        half16 g3 = load_fragB(b3 + k0 + 32);
        if (k0 + 64 < DFF) {
            __builtin_prefetch(b0 + k0 + 64, 0, 1);
            __builtin_prefetch(b2 + k0 + 64, 0, 1);
        }
        acc0 = __builtin_amdgcn_wmma_f32_16x16x32_f16(false, a, false, f0, (short)0, acc0, false, false);
        acc1 = __builtin_amdgcn_wmma_f32_16x16x32_f16(false, a, false, f1, (short)0, acc1, false, false);
        acc2 = __builtin_amdgcn_wmma_f32_16x16x32_f16(false, a, false, f2, (short)0, acc2, false, false);
        acc3 = __builtin_amdgcn_wmma_f32_16x16x32_f16(false, a, false, f3, (short)0, acc3, false, false);
        a = an; f0 = g0; f1 = g1; f2 = g2; f3 = g3;
    }
    acc0 = __builtin_amdgcn_wmma_f32_16x16x32_f16(false, a, false, f0, (short)0, acc0, false, false);
    acc1 = __builtin_amdgcn_wmma_f32_16x16x32_f16(false, a, false, f1, (short)0, acc1, false, false);
    acc2 = __builtin_amdgcn_wmma_f32_16x16x32_f16(false, a, false, f2, (short)0, acc2, false, false);
    acc3 = __builtin_amdgcn_wmma_f32_16x16x32_f16(false, a, false, f3, (short)0, acc3, false, false);

#pragma unroll
    for (int r = 0; r < 8; ++r) {
        int row = row0 + r + khalf * 8;
        if (row < cnt) {
            int   tk = tokIdx[e * T_TOK + row];
            int   sl = tokSlot[e * T_TOK + row];
            float wg = tokWgt[e * T_TOK + row];
            long base = ((long)tk * TOPK + sl) * DM + n0 + lrow;
            eo[base +  0] = wg * acc0[r];
            eo[base + 16] = wg * acc1[r];
            eo[base + 32] = wg * acc2[r];
            eo[base + 48] = wg * acc3[r];
        }
    }
}

// ---------------- kernel 5: combine slots + lb_loss ------------------------
__global__ void moe_combine_kernel(const float* __restrict__ eo,
                                   const float* __restrict__ probsum,
                                   float* __restrict__ out) {
    long gid = (long)blockIdx.x * blockDim.x + threadIdx.x;   // T_TOK*DM threads
    long t = gid / DM;
    long d = gid - t * DM;
    out[gid] = eo[(t * TOPK + 0) * DM + d] + eo[(t * TOPK + 1) * DM + d];
    if (gid == 0) {
        float lb = 0.0f;
#pragma unroll
        for (int e = 0; e < NE; ++e) {
            float m = probsum[e] / (float)T_TOK;
            lb += m * m;
        }
        out[(long)T_TOK * DM] = (float)NE * lb;
    }
}

// ---------------------------------------------------------------------------
extern "C" void kernel_launch(void* const* d_in, const int* in_sizes, int n_in,
                              void* d_out, int out_size, void* d_ws, size_t ws_size,
                              hipStream_t stream) {
    const float* x  = (const float*)d_in[0];   // [T, DM]
    const float* gw = (const float*)d_in[1];   // [NE, DM]
    const float* w1 = (const float*)d_in[2];   // [NE, DFF, DM]
    const float* w2 = (const float*)d_in[3];   // [NE, DM, DFF]
    float* out = (float*)d_out;                // y [T,DM] then lb_loss scalar
    (void)in_sizes; (void)n_in; (void)out_size; (void)ws_size;

    char* ws = (char*)d_ws;
    size_t off = 0;
    _Float16* xh  = (_Float16*)(ws + off); off += (size_t)T_TOK * DM * 2;        // 8 MiB
    _Float16* w1h = (_Float16*)(ws + off); off += (size_t)NE * DFF * DM * 2;     // 64 MiB
    _Float16* w2h = (_Float16*)(ws + off); off += (size_t)NE * DM * DFF * 2;     // 64 MiB
    _Float16* h   = (_Float16*)(ws + off); off += (size_t)T_TOK * TOPK * DFF * 2;// 64 MiB
    float*    eo  = (float*)(ws + off);    off += (size_t)T_TOK * TOPK * DM * 4; // 32 MiB
    int*   counts  = (int*)(ws + off);     off += 64;
    float* probsum = (float*)(ws + off);   off += 64;
    int*   tokIdx  = (int*)(ws + off);     off += (size_t)NE * T_TOK * 4;
    int*   tokSlot = (int*)(ws + off);     off += (size_t)NE * T_TOK * 4;
    float* tokWgt  = (float*)(ws + off);   off += (size_t)NE * T_TOK * 4;

    moe_zero_kernel<<<1, 32, 0, stream>>>(counts, probsum);
    moe_cvt_kernel<<<4096, 256, 0, stream>>>(x,  xh,  (long)T_TOK * DM);
    moe_cvt_kernel<<<8192, 256, 0, stream>>>(w1, w1h, (long)NE * DFF * DM);
    moe_cvt_kernel<<<8192, 256, 0, stream>>>(w2, w2h, (long)NE * DM * DFF);
    moe_gating_kernel<<<T_TOK, 256, 0, stream>>>(x, gw, counts, probsum, tokIdx, tokSlot, tokWgt);
    moe_gemm1_kernel<<<dim3(DFF / 64, T_TOK / 64, NE), 128, 0, stream>>>(
        xh, w1h, h, counts, tokIdx, tokSlot);
    moe_gemm2_kernel<<<dim3(DM / 64, T_TOK / 64, NE), 128, 0, stream>>>(
        h, w2h, eo, counts, tokIdx, tokSlot, tokWgt);
    moe_combine_kernel<<<(T_TOK * DM) / 256, 256, 0, stream>>>(eo, probsum, out);
}
